// EncoderLayer_54674933678318
// MI455X (gfx1250) — matrix operand order repaired and checked by reading, not compiled, not run on hardware
//
#include <hip/hip_runtime.h>
#include <hip/hip_bf16.h>

typedef _Float16 h16;
typedef __attribute__((ext_vector_type(8)))  _Float16 v8h;
typedef __attribute__((ext_vector_type(16))) _Float16 v16h;
typedef __attribute__((ext_vector_type(8)))  float    v8f;

#define WMMA_F16(a, b, c) __builtin_amdgcn_wmma_f32_16x16x32_f16( \
    false, (a), false, (b), (short)0, (c), false, false)

// Two contiguous 16-byte loads -> one 16-half fragment (2x b128 loads)
__device__ __forceinline__ v16h ld2x8(const h16* p0, const h16* p1) {
    v8h lo = *(const v8h*)p0;
    v8h hi = *(const v8h*)p1;
    return __builtin_shufflevector(lo, hi, 0, 1, 2, 3, 4, 5, 6, 7,
                                   8, 9, 10, 11, 12, 13, 14, 15);
}

// A fragment: 16x32 f16 row-major (ISA 7.12.2).
__device__ __forceinline__ v16h load_a_frag(const h16* __restrict__ base, int stride) {
    const int lane = threadIdx.x & 31;
    const h16* p = base + (lane & 15) * stride + ((lane >> 4) << 3);
    return ld2x8(p, p + 16);
}

// B fragment from TRANSPOSED storage Bt[N][K] (row n contiguous along K).
__device__ __forceinline__ v16h load_bt_frag(const h16* __restrict__ base, int stride) {
    const int lane = threadIdx.x & 31;
    const h16* p = base + (lane & 15) * stride + ((lane >> 4) << 3);
    return ld2x8(p, p + 16);
}

// ---------------------------------------------------------------------------
// Tiled WMMA GEMM: C[M,N] = A[M,K] @ B[K,N] + bias, B supplied transposed
// as Bt[N][K]. 256 threads (8 waves), tile 128x128, Kc=32, register-staged
// 2-stage pipeline (issue next chunk's global loads before the WMMA block).
// MODE: 0 = f32 out, 1 = f16 out, 2 = f16 + ReLU,
//       3 = f16 scattered as vT[b][h][d][t], 4 = f16 scaled by log2e/sqrt(dk)
template <int MODE>
__global__ __launch_bounds__(256) void gemm_wmma_kernel(
    const h16* __restrict__ A, const h16* __restrict__ Bt,
    const float* __restrict__ bias, void* __restrict__ Cout,
    int M, int N, int K)
{
    __shared__ __align__(16) h16 As[128 * 48];
    __shared__ __align__(16) h16 Bs[128 * 48];

    const int tid  = threadIdx.x;
    const int lane = tid & 31;
    const int wave = tid >> 5;
    const int wm = wave & 3, wn = wave >> 2;
    const int m0 = blockIdx.y * 128, n0 = blockIdx.x * 128;

    int rowt[2], c4t[2];
#pragma unroll
    for (int t = 0; t < 2; ++t) {
        const int idx = tid + t * 256;
        rowt[t] = idx >> 2;
        c4t[t]  = idx & 3;
    }

    v8f acc[2][4];
#pragma unroll
    for (int mi = 0; mi < 2; ++mi)
#pragma unroll
        for (int ni = 0; ni < 4; ++ni)
#pragma unroll
            for (int e = 0; e < 8; ++e) acc[mi][ni][e] = 0.0f;

    uint4 ra[2], rb[2];
#pragma unroll
    for (int t = 0; t < 2; ++t) {
        ra[t] = *(const uint4*)(A  + (size_t)(m0 + rowt[t]) * K + c4t[t] * 8);
        rb[t] = *(const uint4*)(Bt + (size_t)(n0 + rowt[t]) * K + c4t[t] * 8);
    }

    for (int k0 = 0; k0 < K; k0 += 32) {
        // commit staged chunk to LDS
#pragma unroll
        for (int t = 0; t < 2; ++t) {
            *(uint4*)&As[rowt[t] * 48 + c4t[t] * 8] = ra[t];
            *(uint4*)&Bs[rowt[t] * 48 + c4t[t] * 8] = rb[t];
        }
        __syncthreads();
        // issue next chunk's global loads; latency hidden behind the WMMAs
        if (k0 + 32 < K) {
#pragma unroll
            for (int t = 0; t < 2; ++t) {
                ra[t] = *(const uint4*)(A  + (size_t)(m0 + rowt[t]) * K + k0 + 32 + c4t[t] * 8);
                rb[t] = *(const uint4*)(Bt + (size_t)(n0 + rowt[t]) * K + k0 + 32 + c4t[t] * 8);
            }
        }

        v16h af[2], bf[4];
#pragma unroll
        for (int mi = 0; mi < 2; ++mi)
            af[mi] = load_a_frag(&As[(wm * 32 + mi * 16) * 48], 48);
#pragma unroll
        for (int ni = 0; ni < 4; ++ni)
            bf[ni] = load_bt_frag(&Bs[(wn * 64 + ni * 16) * 48], 48);
#pragma unroll
        for (int mi = 0; mi < 2; ++mi)
#pragma unroll
            for (int ni = 0; ni < 4; ++ni)
                acc[mi][ni] = WMMA_F16(af[mi], bf[ni], acc[mi][ni]);
        __syncthreads();
    }

    // Epilogue (C layout: lane<16 -> row r, col lane; lane>=16 -> row 8+r)
    const int nl = lane & 15, moff = (lane >> 4) * 8;
    const float QSCALE = 0.125f * 1.44269504088896340736f;  // log2e / sqrt(64)
#pragma unroll
    for (int mi = 0; mi < 2; ++mi) {
#pragma unroll
        for (int ni = 0; ni < 4; ++ni) {
            const int row = m0 + wm * 32 + mi * 16 + moff;
            const int col = n0 + wn * 64 + ni * 16 + nl;
            const float bv = bias[col];
#pragma unroll
            for (int r = 0; r < 8; ++r) {
                float vv = acc[mi][ni][r] + bv;
                if (MODE == 2) vv = fmaxf(vv, 0.0f);
                if (MODE == 4) vv *= QSCALE;
                if (MODE == 0) {
                    ((float*)Cout)[(size_t)(row + r) * N + col] = vv;
                } else if (MODE == 3) {
                    const int rr = row + r;
                    const size_t o = (((size_t)((rr >> 11) * 16 + (col >> 6)) * 64
                                       + (col & 63)) << 11) + (rr & 2047);
                    ((h16*)Cout)[o] = (h16)vv;
                } else {
                    ((h16*)Cout)[(size_t)(row + r) * N + col] = (h16)vv;
                }
            }
        }
    }
}

// ---------------------------------------------------------------------------
// Flash-style attention; q is pre-scaled by log2e/sqrt(dk) so the online
// softmax runs in the exp2 domain. One block per (b, h, 128-query tile);
// each of 8 waves owns 16 full query rows. K/V tiles register-staged.
__global__ __launch_bounds__(256) void attn_wmma_kernel(
    const h16* __restrict__ q, const h16* __restrict__ k,
    const h16* __restrict__ vT, h16* __restrict__ o)
{
    constexpr int T = 2048, DM = 1024;
    // region0: KT tile 128x72 (9216) aliased with P slabs 8*16*136 (17408)
    // region1: VT tile 64x136 (8704)
    __shared__ __align__(16) h16 smem[17408 + 8704];
    h16* KT = smem;
    h16* VT = smem + 17408;

    const int tid  = threadIdx.x;
    const int lane = tid & 31;
    const int wave = tid >> 5;
    const int qt = blockIdx.x, hh = blockIdx.y, bb = blockIdx.z;
    const size_t headoff = (size_t)hh * 64;
    const int qr0 = qt * 128 + wave * 16;
    const int nl = lane & 15, moff = (lane >> 4) * 8;
    const h16* vhead = vT + ((size_t)(bb * 16 + hh) << 17);  // *64*2048

    int krow[4], kc8[4], vdr[4], vc[4];
#pragma unroll
    for (int t = 0; t < 4; ++t) {
        const int idx = tid + t * 256;
        krow[t] = idx >> 3; kc8[t] = idx & 7;     // K tile: 128 rows x 8 chunks
        vdr[t]  = idx >> 4; vc[t]  = idx & 15;    // V tile: 64 rows x 16 chunks
    }

    // Q fragments (16 rows x 64 = 2 K-chunks), straight from global
    v16h qf[2];
#pragma unroll
    for (int kc = 0; kc < 2; ++kc)
        qf[kc] = load_a_frag(q + (size_t)(bb * T + qr0) * DM + headoff + kc * 32, DM);

    v8f oacc[4];
#pragma unroll
    for (int dt = 0; dt < 4; ++dt)
#pragma unroll
        for (int e = 0; e < 8; ++e) oacc[dt][e] = 0.0f;
    float mrow[8], lrow[8];
#pragma unroll
    for (int r = 0; r < 8; ++r) { mrow[r] = -3.0e38f; lrow[r] = 0.0f; }

    // stage first K/V tiles
    uint4 rk[4], rv[4];
#pragma unroll
    for (int t = 0; t < 4; ++t) {
        rk[t] = *(const uint4*)(k + (size_t)(bb * T + krow[t]) * DM + headoff + kc8[t] * 8);
        rv[t] = *(const uint4*)(vhead + ((size_t)vdr[t] << 11) + vc[t] * 8);
    }

    for (int kt = 0; kt < T; kt += 128) {
        // commit staged tiles: KT[key][d] stride 72, VT[d][key] stride 136
#pragma unroll
        for (int t = 0; t < 4; ++t) {
            *(uint4*)&KT[krow[t] * 72 + kc8[t] * 8] = rk[t];
            *(uint4*)&VT[vdr[t] * 136 + vc[t] * 8]  = rv[t];
        }
        __syncthreads();
        // issue next tile's global loads; hidden behind S/softmax/PV
        if (kt + 128 < T) {
#pragma unroll
            for (int t = 0; t < 4; ++t) {
                rk[t] = *(const uint4*)(k + (size_t)(bb * T + kt + 128 + krow[t]) * DM
                                        + headoff + kc8[t] * 8);
                rv[t] = *(const uint4*)(vhead + ((size_t)vdr[t] << 11) + kt + 128 + vc[t] * 8);
            }
        }

        // S = Q @ K^T : logical B = [d][key]; its [key][d] layout == KT
        v8f sacc[8];
#pragma unroll
        for (int nt = 0; nt < 8; ++nt) {
#pragma unroll
            for (int e = 0; e < 8; ++e) sacc[nt][e] = 0.0f;
#pragma unroll
            for (int kc = 0; kc < 2; ++kc)
                sacc[nt] = WMMA_F16(qf[kc],
                                    load_bt_frag(&KT[(nt * 16) * 72 + kc * 32], 72),
                                    sacc[nt]);
        }
        __syncthreads();  // all waves done reading KT; safe to overwrite with P

        // Online softmax (exp2 domain); probabilities into per-wave P slab
        h16* P = KT + wave * 16 * 136;
#pragma unroll
        for (int r = 0; r < 8; ++r) {
            float mx = -3.0e38f;
#pragma unroll
            for (int nt = 0; nt < 8; ++nt) mx = fmaxf(mx, sacc[nt][r]);
#pragma unroll
            for (int d = 1; d < 16; d <<= 1) mx = fmaxf(mx, __shfl_xor(mx, d, 32));
            const float mnew = fmaxf(mrow[r], mx);
            const float corr = exp2f(mrow[r] - mnew);
            float ps = 0.0f;
#pragma unroll
            for (int nt = 0; nt < 8; ++nt) {
                float p = exp2f(sacc[nt][r] - mnew);
                ps += p;
                P[(moff + r) * 136 + nt * 16 + nl] = (h16)p;
            }
#pragma unroll
            for (int d = 1; d < 16; d <<= 1) ps += __shfl_xor(ps, d, 32);
            lrow[r] = lrow[r] * corr + ps;
            mrow[r] = mnew;
#pragma unroll
            for (int dt = 0; dt < 4; ++dt) oacc[dt][r] *= corr;
        }

        // O += P @ V : logical B = [key][d]; its [d][key] layout == VT
#pragma unroll
        for (int dt = 0; dt < 4; ++dt)
#pragma unroll
            for (int kc = 0; kc < 4; ++kc) {
                v16h pa = load_a_frag(P + kc * 32, 136);
                v16h vb = load_bt_frag(&VT[(dt * 16) * 136 + kc * 32], 136);
                oacc[dt] = WMMA_F16(pa, vb, oacc[dt]);
            }
        __syncthreads();  // before next iteration overwrites KT/VT
    }

    // Normalize and write out (f16, [B*T, 1024])
    float rinv[8];
#pragma unroll
    for (int r = 0; r < 8; ++r) rinv[r] = 1.0f / lrow[r];
#pragma unroll
    for (int dt = 0; dt < 4; ++dt) {
#pragma unroll
        for (int r = 0; r < 8; ++r) {
            const int row = qr0 + moff + r;
            o[(size_t)(bb * T + row) * DM + headoff + dt * 16 + nl] =
                (h16)(oacc[dt][r] * rinv[r]);
        }
    }
}

// ---------------------------------------------------------------------------
// Fused residual + LayerNorm over D=1024; one row per block of 256 threads.
__global__ __launch_bounds__(256) void ln_residual_kernel(
    const float* __restrict__ a, const float* __restrict__ res,
    const float* __restrict__ gamma, const float* __restrict__ beta,
    float* __restrict__ y, h16* __restrict__ yh)
{
    __shared__ float red[256];
    const int row = blockIdx.x;
    const size_t base = (size_t)row * 1024;
    float vals[4];
    float s = 0.0f;
#pragma unroll
    for (int i = 0; i < 4; ++i) {
        const int idx = threadIdx.x + i * 256;
        const float t = a[base + idx] + res[base + idx];
        vals[i] = t;
        s += t;
    }
    red[threadIdx.x] = s;
    __syncthreads();
    for (int off = 128; off > 0; off >>= 1) {
        if (threadIdx.x < off) red[threadIdx.x] += red[threadIdx.x + off];
        __syncthreads();
    }
    const float mean = red[0] * (1.0f / 1024.0f);
    __syncthreads();
    float sq = 0.0f;
#pragma unroll
    for (int i = 0; i < 4; ++i) {
        const float d = vals[i] - mean;
        sq += d * d;
    }
    red[threadIdx.x] = sq;
    __syncthreads();
    for (int off = 128; off > 0; off >>= 1) {
        if (threadIdx.x < off) red[threadIdx.x] += red[threadIdx.x + off];
        __syncthreads();
    }
    const float rinv = rsqrtf(red[0] * (1.0f / 1024.0f) + 1e-12f);
#pragma unroll
    for (int i = 0; i < 4; ++i) {
        const int idx = threadIdx.x + i * 256;
        const float yv = gamma[idx] * (vals[i] - mean) * rinv + beta[idx];
        y[base + idx] = yv;
        if (yh) yh[base + idx] = (h16)yv;
    }
}

// f32 -> f16 elementwise conversion (for x)
__global__ __launch_bounds__(256) void cvt_f32_f16_kernel(
    const float* __restrict__ src, h16* __restrict__ dst, int n)
{
    const int i = blockIdx.x * 256 + threadIdx.x;
    if (i < n) dst[i] = (h16)src[i];
}

// f32 [K][N] -> f16 transposed [N][K], tiled through LDS
__global__ void cvt_transpose_kernel(const float* __restrict__ src,
                                     h16* __restrict__ dst, int K, int N)
{
    __shared__ float tile[32][33];
    const int gx = blockIdx.x * 32;  // N direction
    const int gy = blockIdx.y * 32;  // K direction
#pragma unroll
    for (int j = 0; j < 4; ++j)
        tile[threadIdx.y + j * 8][threadIdx.x] =
            src[(size_t)(gy + threadIdx.y + j * 8) * N + gx + threadIdx.x];
    __syncthreads();
#pragma unroll
    for (int j = 0; j < 4; ++j)
        dst[(size_t)(gx + threadIdx.y + j * 8) * K + gy + threadIdx.x] =
            (h16)tile[threadIdx.x][threadIdx.y + j * 8];
}

// ---------------------------------------------------------------------------
extern "C" void kernel_launch(void* const* d_in, const int* in_sizes, int n_in,
                              void* d_out, int out_size, void* d_ws, size_t ws_size,
                              hipStream_t stream) {
    (void)in_sizes; (void)n_in; (void)out_size; (void)ws_size;
    constexpr int B = 4, T = 2048, D = 1024, F = 4096;
    constexpr int BT = B * T;

    const float* x     = (const float*)d_in[0];
    const float* w_q   = (const float*)d_in[1];
    const float* b_q   = (const float*)d_in[2];
    const float* w_k   = (const float*)d_in[3];
    const float* b_k   = (const float*)d_in[4];
    const float* w_v   = (const float*)d_in[5];
    const float* b_v   = (const float*)d_in[6];
    const float* w_o   = (const float*)d_in[7];
    const float* b_o   = (const float*)d_in[8];
    const float* gamma1= (const float*)d_in[9];
    const float* beta1 = (const float*)d_in[10];
    const float* w_fc1 = (const float*)d_in[11];
    const float* b_fc1 = (const float*)d_in[12];
    const float* w_fc2 = (const float*)d_in[13];
    const float* b_fc2 = (const float*)d_in[14];
    const float* gamma2= (const float*)d_in[15];
    const float* beta2 = (const float*)d_in[16];

    char* ws = (char*)d_ws;
    size_t off = 0;
    auto alloc = [&](size_t bytes) -> void* {
        void* p = ws + off;
        off += (bytes + 255) & ~(size_t)255;
        return p;
    };
    h16*   xh   = (h16*)  alloc((size_t)BT * D * 2);
    h16*   wqt  = (h16*)  alloc((size_t)D * D * 2);   // transposed weights [N][K]
    h16*   wkt  = (h16*)  alloc((size_t)D * D * 2);
    h16*   wvt  = (h16*)  alloc((size_t)D * D * 2);
    h16*   wot  = (h16*)  alloc((size_t)D * D * 2);
    h16*   w1t  = (h16*)  alloc((size_t)D * F * 2);   // [4096][1024]
    h16*   w2t  = (h16*)  alloc((size_t)F * D * 2);   // [1024][4096]
    h16*   qb   = (h16*)  alloc((size_t)BT * D * 2);
    h16*   kb   = (h16*)  alloc((size_t)BT * D * 2);
    h16*   vTb  = (h16*)  alloc((size_t)BT * D * 2);  // vT[b][h][64][2048]
    h16*   ob   = (h16*)  alloc((size_t)BT * D * 2);
    float* attn = (float*)alloc((size_t)BT * D * 4);
    float* x1   = (float*)alloc((size_t)BT * D * 4);
    h16*   x1h  = (h16*)  alloc((size_t)BT * D * 2);
    h16*   hb   = (h16*)  alloc((size_t)BT * F * 2);
    float* ffn  = (float*)alloc((size_t)BT * D * 4);

    cvt_f32_f16_kernel<<<(BT * D + 255) / 256, 256, 0, stream>>>(x, xh, BT * D);
    const dim3 tb(32, 8);
    cvt_transpose_kernel<<<dim3(D / 32, D / 32), tb, 0, stream>>>(w_q,   wqt, D, D);
    cvt_transpose_kernel<<<dim3(D / 32, D / 32), tb, 0, stream>>>(w_k,   wkt, D, D);
    cvt_transpose_kernel<<<dim3(D / 32, D / 32), tb, 0, stream>>>(w_v,   wvt, D, D);
    cvt_transpose_kernel<<<dim3(D / 32, D / 32), tb, 0, stream>>>(w_o,   wot, D, D);
    cvt_transpose_kernel<<<dim3(F / 32, D / 32), tb, 0, stream>>>(w_fc1, w1t, D, F);
    cvt_transpose_kernel<<<dim3(D / 32, F / 32), tb, 0, stream>>>(w_fc2, w2t, F, D);

    const dim3 blk(256);
    const dim3 gDD(D / 128, BT / 128);
    const dim3 gDF(F / 128, BT / 128);

    // QKV projections (q pre-scaled by log2e/sqrt(dk) for exp2-domain softmax)
    gemm_wmma_kernel<4><<<gDD, blk, 0, stream>>>(xh, wqt, b_q, qb,  BT, D, D);
    gemm_wmma_kernel<1><<<gDD, blk, 0, stream>>>(xh, wkt, b_k, kb,  BT, D, D);
    gemm_wmma_kernel<3><<<gDD, blk, 0, stream>>>(xh, wvt, b_v, vTb, BT, D, D);

    // Multi-head attention
    attn_wmma_kernel<<<dim3(T / 128, 16, B), blk, 0, stream>>>(qb, kb, vTb, ob);

    // Output projection (f32 out for residual)
    gemm_wmma_kernel<0><<<gDD, blk, 0, stream>>>(ob, wot, b_o, attn, BT, D, D);

    // LN1: (attn + x) -> x1 (f32) and x1h (f16)
    ln_residual_kernel<<<BT, blk, 0, stream>>>(attn, x, gamma1, beta1, x1, x1h);

    // FFN
    gemm_wmma_kernel<2><<<gDF, blk, 0, stream>>>(x1h, w1t, b_fc1, hb, BT, F, D);
    gemm_wmma_kernel<0><<<gDD, blk, 0, stream>>>(hb, w2t, b_fc2, ffn, BT, D, F);

    // LN2: (ffn + x1) -> d_out (f32)
    ln_residual_kernel<<<BT, blk, 0, stream>>>(ffn, x1, gamma2, beta2, (float*)d_out, nullptr);
}
